// AttentiveAtlasEncoder_27195732918748
// MI455X (gfx1250) — matrix operand order repaired and checked by reading, not compiled, not run on hardware
//
#include <hip/hip_runtime.h>
#include <math.h>

typedef __attribute__((ext_vector_type(16))) _Float16 v16h;
typedef __attribute__((ext_vector_type(8)))  float    v8f;

#define BATCH 1000000
#define HID   32
#define NCH   3
#define NKC   21

// LDS layout (floats)
#define OFF_W1   0       // 2x32
#define OFF_B1   64      // 32
#define OFF_W2   96      // 32x32
#define OFF_B2   1120    // 32
#define OFF_VW   1152    // 32x2
#define OFF_VB   1216    // 2
#define OFF_CC   1218    // 3x2
#define OFF_CB   1224    // 3x21x2
#define OFF_SW1  1350    // 2x2
#define OFF_SB1  1354    // 2
#define OFF_SW2  1356    // 2x2
#define OFF_SB2  1360    // 2
#define OFF_H2   1362    // 8 waves * 1056 (2 tiles * 16 rows * stride 33)
#define OFF_RED  (OFF_H2 + 8 * 1056)
#define SMEM_FLOATS (OFF_RED + 8)

__device__ __forceinline__ float gelu_exact(float x) {
    return 0.5f * x * (1.0f + erff(x * 0.70710678118654752f));
}

__global__ __launch_bounds__(256)
void atlas_encoder_kernel(const float* __restrict__ x,
                          const float* __restrict__ feat_w1,
                          const float* __restrict__ feat_b1,
                          const float* __restrict__ feat_w2,
                          const float* __restrict__ feat_b2,
                          const float* __restrict__ val_w,
                          const float* __restrict__ val_b,
                          const float* __restrict__ chart_centers,
                          const float* __restrict__ codebook,
                          const float* __restrict__ sf_w1,
                          const float* __restrict__ sf_b1,
                          const float* __restrict__ sf_w2,
                          const float* __restrict__ sf_b2,
                          float* __restrict__ out,
                          float* __restrict__ block_partials)
{
    __shared__ float sm[SMEM_FLOATS];

    const int tid  = threadIdx.x;
    const int wave = tid >> 5;
    const int L    = tid & 31;   // lane in wave32
    const int ml   = L & 15;     // M within tile / N within tile
    const int kg   = L >> 4;     // K-group selector per ISA A/B layouts

    const size_t Bn = BATCH;
    const long blockStart = (long)blockIdx.x * 256;
    const long waveBase   = blockStart + (long)wave * 32;

    // warm L0/L2 for this wave's x slice while we stage weights in LDS
    {
        long ps = waveBase + L;
        if (ps >= (long)Bn) ps = (long)Bn - 1;
        __builtin_prefetch(&x[2 * ps], 0, 1);
    }

    // ---- stage all weights into LDS ----
    for (int i = tid; i < 64;   i += 256) sm[OFF_W1 + i] = feat_w1[i];
    for (int i = tid; i < 32;   i += 256) sm[OFF_B1 + i] = feat_b1[i];
    for (int i = tid; i < 1024; i += 256) sm[OFF_W2 + i] = feat_w2[i];
    for (int i = tid; i < 32;   i += 256) sm[OFF_B2 + i] = feat_b2[i];
    for (int i = tid; i < 64;   i += 256) sm[OFF_VW + i] = val_w[i];
    for (int i = tid; i < 2;    i += 256) sm[OFF_VB + i] = val_b[i];
    for (int i = tid; i < 6;    i += 256) sm[OFF_CC + i] = chart_centers[i];
    for (int i = tid; i < NCH*NKC*2; i += 256) sm[OFF_CB + i] = codebook[i];
    for (int i = tid; i < 4;    i += 256) sm[OFF_SW1 + i] = sf_w1[i];
    for (int i = tid; i < 2;    i += 256) sm[OFF_SB1 + i] = sf_b1[i];
    for (int i = tid; i < 4;    i += 256) sm[OFF_SW2 + i] = sf_w2[i];
    for (int i = tid; i < 2;    i += 256) sm[OFF_SB2 + i] = sf_b2[i];
    __syncthreads();

    // ---- build split-f16 B fragments for feat_w2 (32x16 per fragment) ----
    // B k order (this code's, same as A): lanes 0-15 hold K=0..15 (2 per vgpr), lanes 16-31 K=16..31; N = lane%16
    // Split scheme: w = hi + lo with hi = f16(w), lo = f16(w - hi)  (~21-bit mantissa)
    v16h bh0, bl0, bh1, bl1;
    {
        const float* w2 = &sm[OFF_W2];
        const int kb = kg * 16;
        #pragma unroll
        for (int v = 0; v < 16; ++v) {
            float w0 = w2[(kb + v) * 32 + ml];
            float w1v = w2[(kb + v) * 32 + 16 + ml];
            _Float16 h0 = (_Float16)w0;
            _Float16 h1v = (_Float16)w1v;
            bh0[v] = h0;  bl0[v] = (_Float16)(w0  - (float)h0);
            bh1[v] = h1v; bl1[v] = (_Float16)(w1v - (float)h1v);
        }
    }

    // ---- per-tile: layer1 (scalar, directly into A-fragment layout), WMMA layer2 ----
    float* h2wave = &sm[OFF_H2 + wave * 1056];
    #pragma unroll
    for (int t = 0; t < 2; ++t) {
        long s  = waveBase + t * 16 + ml;
        long sc = s < (long)Bn ? s : (long)Bn - 1;   // clamp: keep EXEC all-ones
        float2 xv = *(const float2*)(&x[2 * sc]);

        // A fragment: halves i -> K = 16*kg + i; split hi/lo
        v16h ah, al;
        #pragma unroll
        for (int i = 0; i < 16; ++i) {
            int j = kg * 16 + i;
            float hv = xv.x * sm[OFF_W1 + j] + xv.y * sm[OFF_W1 + 32 + j] + sm[OFF_B1 + j];
            float g  = gelu_exact(hv);
            _Float16 gh = (_Float16)g;
            ah[i] = gh;
            al[i] = (_Float16)(g - (float)gh);
        }

        // D = Al*Bh + (Ah*Bl + (Ah*Bh + 0))  -> near-f32 accuracy, f32 accumulation
        v8f cz = {};
        v8f acc0 = __builtin_amdgcn_wmma_f32_16x16x32_f16(false, ah, false, bh0,
                                                          (short)0, cz, false, false);
        acc0 = __builtin_amdgcn_wmma_f32_16x16x32_f16(false, ah, false, bl0,
                                                      (short)0, acc0, false, false);
        acc0 = __builtin_amdgcn_wmma_f32_16x16x32_f16(false, al, false, bh0,
                                                      (short)0, acc0, false, false);

        v8f acc1 = __builtin_amdgcn_wmma_f32_16x16x32_f16(false, ah, false, bh1,
                                                          (short)0, cz, false, false);
        acc1 = __builtin_amdgcn_wmma_f32_16x16x32_f16(false, ah, false, bl1,
                                                      (short)0, acc1, false, false);
        acc1 = __builtin_amdgcn_wmma_f32_16x16x32_f16(false, al, false, bh1,
                                                      (short)0, acc1, false, false);

        // bias + gelu, store h2 tile to LDS (C layout: vgpr r -> M = r + 8*kg, N = ml)
        float* h2t = h2wave + t * 528;
        #pragma unroll
        for (int r = 0; r < 8; ++r) {
            int mr = r + 8 * kg;
            h2t[mr * 33 + ml]      = gelu_exact(acc0[r] + sm[OFF_B2 + ml]);
            h2t[mr * 33 + 16 + ml] = gelu_exact(acc1[r] + sm[OFF_B2 + 16 + ml]);
        }
    }
    __syncthreads();

    // ---- scalar tail: one lane per sample ----
    const long s = waveBase + L;
    const bool act = (s < (long)Bn);
    const float* h2row = &sm[OFF_H2 + wave * 1056 + (L >> 4) * 528 + ml * 33];

    // v = h2 @ val_w + val_b
    float v0 = sm[OFF_VB + 0], v1 = sm[OFF_VB + 1];
    #pragma unroll
    for (int j = 0; j < HID; ++j) {
        float hv = h2row[j];
        v0 = fmaf(hv, sm[OFF_VW + 2*j],     v0);
        v1 = fmaf(hv, sm[OFF_VW + 2*j + 1], v1);
    }

    // router = softmax(v @ centers^T / sqrt(D)), K_chart = argmax
    float sc0[NCH];
    #pragma unroll
    for (int c = 0; c < NCH; ++c)
        sc0[c] = (v0 * sm[OFF_CC + 2*c] + v1 * sm[OFF_CC + 2*c + 1]) * 0.70710678118654752f;
    float mx = sc0[0]; int kchart = 0;
    #pragma unroll
    for (int c = 1; c < NCH; ++c) if (sc0[c] > mx) { mx = sc0[c]; kchart = c; }
    float e[NCH], esum = 0.f;
    #pragma unroll
    for (int c = 0; c < NCH; ++c) { e[c] = __expf(sc0[c] - mx); esum += e[c]; }
    float r[NCH];
    float inv = 1.f / esum;
    #pragma unroll
    for (int c = 0; c < NCH; ++c) r[c] = e[c] * inv;

    // c_bar, v_local
    float cb0 = 0.f, cb1 = 0.f;
    #pragma unroll
    for (int c = 0; c < NCH; ++c) {
        cb0 = fmaf(r[c], sm[OFF_CC + 2*c],     cb0);
        cb1 = fmaf(r[c], sm[OFF_CC + 2*c + 1], cb1);
    }
    float vl0 = v0 - cb0, vl1 = v1 - cb1;

    // per-chart nearest code, z_q, loss, sf-MLP
    int   idx[NCH];
    float zn0c[NCH], zn1c[NCH];
    float zb0 = 0.f, zb1 = 0.f, zn0 = 0.f, zn1 = 0.f, lossv = 0.f;
    #pragma unroll
    for (int c = 0; c < NCH; ++c) {
        const float* cbk = &sm[OFF_CB + c * NKC * 2];
        float bestD = 3.4e38f; int bestK = 0;
        for (int k = 0; k < NKC; ++k) {
            float d0 = vl0 - cbk[2*k], d1 = vl1 - cbk[2*k + 1];
            float dd = d0 * d0 + d1 * d1;
            if (dd < bestD) { bestD = dd; bestK = k; }
        }
        idx[c] = bestK;
        float q0 = cbk[2*bestK], q1 = cbk[2*bestK + 1];
        lossv = fmaf(r[c], bestD, lossv);                 // (v_local - z_q)^2 summed over D
        zb0 = fmaf(r[c], q0, zb0);
        zb1 = fmaf(r[c], q1, zb1);
        // shading-field MLP on delta
        float d0 = vl0 - q0, d1 = vl1 - q1;
        float t0 = gelu_exact(d0 * sm[OFF_SW1+0] + d1 * sm[OFF_SW1+2] + sm[OFF_SB1+0]);
        float t1 = gelu_exact(d0 * sm[OFF_SW1+1] + d1 * sm[OFF_SW1+3] + sm[OFF_SB1+1]);
        float z0 = t0 * sm[OFF_SW2+0] + t1 * sm[OFF_SW2+2] + sm[OFF_SB2+0];
        float z1 = t0 * sm[OFF_SW2+1] + t1 * sm[OFF_SW2+3] + sm[OFF_SB2+1];
        zn0c[c] = z0; zn1c[c] = z1;
        zn0 = fmaf(r[c], z0, zn0);
        zn1 = fmaf(r[c], z1, zn1);
    }

    float db0 = vl0 - zb0, db1 = vl1 - zb1;
    float tex0 = db0 - zn0, tex1 = db1 - zn1;
    float geo0 = cb0 + zb0, geo1 = cb1 + zb1;
    int kcode = idx[kchart];

    if (act) {
        size_t ss = (size_t)s;
        out[ss]      = (float)kchart;                            // K_chart
        out[Bn + ss] = (float)kcode;                             // K_code
        *(float2*)(&out[2*Bn + 2*ss]) = make_float2(zn0, zn1);   // z_n   (8B aligned)
        *(float2*)(&out[4*Bn + 2*ss]) = make_float2(tex0, tex1); // z_tex (8B aligned)
        #pragma unroll
        for (int c = 0; c < NCH; ++c) out[6*Bn + 3*ss + c] = r[c];        // router
        *(float2*)(&out[9*Bn + 2*ss]) = make_float2(geo0, geo1); // z_geo (8B aligned)
        #pragma unroll
        for (int c = 0; c < NCH; ++c) out[11*Bn + 1 + 3*ss + c] = (float)idx[c]; // indices
        #pragma unroll
        for (int c = 0; c < NCH; ++c) {                          // z_n_all (odd base: scalar)
            out[14*Bn + 1 + 6*ss + 2*c]     = zn0c[c];
            out[14*Bn + 1 + 6*ss + 2*c + 1] = zn1c[c];
        }
        out[20*Bn + 1 + 2*ss]     = cb0;                         // c_bar (odd base: scalar)
        out[20*Bn + 1 + 2*ss + 1] = cb1;
    }

    // ---- block loss reduction -> per-block partial ----
    float lv = act ? lossv : 0.f;
    #pragma unroll
    for (int off = 16; off > 0; off >>= 1) lv += __shfl_down(lv, off, 32);
    if (L == 0) sm[OFF_RED + wave] = lv;
    __syncthreads();
    if (tid == 0) {
        float tacc = 0.f;
        #pragma unroll
        for (int w = 0; w < 8; ++w) tacc += sm[OFF_RED + w];
        block_partials[blockIdx.x] = tacc;
    }
}

__global__ __launch_bounds__(256)
void atlas_loss_reduce(const float* __restrict__ partials, int n, float* __restrict__ out_slot)
{
    __shared__ float red[256];
    float s = 0.f;
    for (int i = threadIdx.x; i < n; i += 256) s += partials[i];
    red[threadIdx.x] = s;
    __syncthreads();
    for (int off = 128; off > 0; off >>= 1) {
        if (threadIdx.x < off) red[threadIdx.x] += red[threadIdx.x + off];
        __syncthreads();
    }
    if (threadIdx.x == 0) {
        // vq_loss = codebook_loss + 0.25*commitment; both sums are numerically equal
        // => 1.25 * sum / (B * D)
        out_slot[0] = red[0] * (1.25f / ((float)BATCH * 2.0f));
    }
}

extern "C" void kernel_launch(void* const* d_in, const int* in_sizes, int n_in,
                              void* d_out, int out_size, void* d_ws, size_t ws_size,
                              hipStream_t stream) {
    const float* x             = (const float*)d_in[0];
    const float* feat_w1       = (const float*)d_in[1];
    const float* feat_b1       = (const float*)d_in[2];
    const float* feat_w2       = (const float*)d_in[3];
    const float* feat_b2       = (const float*)d_in[4];
    const float* val_w         = (const float*)d_in[5];
    const float* val_b         = (const float*)d_in[6];
    const float* chart_centers = (const float*)d_in[7];
    const float* codebook      = (const float*)d_in[8];
    const float* sf_w1         = (const float*)d_in[9];
    const float* sf_b1         = (const float*)d_in[10];
    const float* sf_w2         = (const float*)d_in[11];
    const float* sf_b2         = (const float*)d_in[12];

    float* out = (float*)d_out;
    float* ws  = (float*)d_ws;

    const int nb = (BATCH + 255) / 256;   // 3907 blocks; B % 32 == 0 so wave tiles are full
    atlas_encoder_kernel<<<nb, 256, 0, stream>>>(
        x, feat_w1, feat_b1, feat_w2, feat_b2, val_w, val_b,
        chart_centers, codebook, sf_w1, sf_b1, sf_w2, sf_b2, out, ws);

    atlas_loss_reduce<<<1, 256, 0, stream>>>(ws, nb, out + (size_t)11 * BATCH);
}
